// SAGEMultiSwitchModel_83408264888625
// MI455X (gfx1250) — matrix operand order repaired and checked by reading, not compiled, not run on hardware
//
#include <hip/hip_runtime.h>

typedef __attribute__((ext_vector_type(16))) _Float16 v16h;
typedef __attribute__((ext_vector_type(8)))  float    v8f;

#define NNODES 100000
#define NEDGES 1600000

// ---------------- WMMA fragment loaders ----------------
// A (16x32 f16): lane = {m = lane&15, hi = lane>>4}; element i=(2v+h):
//   k = kb*32 + (v<4?0:16) + hi*8 + 2*(v&3) + h
__device__ __forceinline__ v16h load_a_f32(const float* __restrict__ row, int kb, int hi, float scale) {
  v16h a;
  const int base = kb * 32 + hi * 8;
#pragma unroll
  for (int v = 0; v < 8; ++v) {
    int k = base + ((v < 4) ? 0 : 16) + 2 * (v & 3);
    float2 f = *(const float2*)(row + k);
    a[2 * v + 0] = (_Float16)(f.x * scale);
    a[2 * v + 1] = (_Float16)(f.y * scale);
  }
  return a;
}

__device__ __forceinline__ v16h load_a_f16(const _Float16* __restrict__ row, int kb, int hi) {
  v16h a;
  const int base = kb * 32 + hi * 8;
#pragma unroll
  for (int v = 0; v < 8; ++v) {
    int k = base + ((v < 4) ? 0 : 16) + 2 * (v & 3);
    a[2 * v + 0] = row[k];
    a[2 * v + 1] = row[k + 1];
  }
  return a;
}

// B (32x16 f16): lane n = lane&15 -> output col j; element i -> k = kb*32 + (i<8?0:16) + (lane>>4)*8 + (i&7)
// => two 8-half chunks of weight row j (weights row-major [out][in]).
__device__ __forceinline__ v16h load_b16(const _Float16* __restrict__ w, int j, int ldw, int kb, int hi) {
  typedef __attribute__((ext_vector_type(8))) _Float16 v8h_t;
  const _Float16* p = w + (size_t)j * ldw + kb * 32 + hi * 8;
  v8h_t lo = *(const v8h_t*)(p);
  v8h_t up = *(const v8h_t*)(p + 16);
  return __builtin_shufflevector(lo, up, 0, 1, 2, 3, 4, 5, 6, 7, 8, 9, 10, 11, 12, 13, 14, 15);
}

// ---------------- utility kernels ----------------
__global__ void zero_f4(float4* __restrict__ p, int n4) {
  int i = blockIdx.x * blockDim.x + threadIdx.x;
  if (i < n4) p[i] = make_float4(0.f, 0.f, 0.f, 0.f);
}

__global__ void convert_w(const float* __restrict__ Wn0, const float* __restrict__ Ws0,
                          const float* __restrict__ Wn1, const float* __restrict__ Ws1,
                          _Float16* __restrict__ w16) {
  int i = blockIdx.x * blockDim.x + threadIdx.x;
  if (i >= 98304) return;
  float v;
  if (i < 16384)      v = Ws0[i];
  else if (i < 32768) v = Wn0[i - 16384];
  else if (i < 65536) v = Ws1[i - 32768];
  else                v = Wn1[i - 65536];
  w16[i] = (_Float16)v;
}

__global__ void deg_kernel(const int* __restrict__ src, const int* __restrict__ dst,
                           float* __restrict__ degO, float* __restrict__ degI) {
  int e = blockIdx.x * blockDim.x + threadIdx.x;
  if (e >= NEDGES) return;
  atomicAdd(&degO[dst[e]], 1.f);   // in-degree ('O': d = dst)
  atomicAdd(&degI[src[e]], 1.f);   // out-degree ('I': d = src)
}

// Layer-0 scatter: x is [N,64]. sumO: 'O' (x[src] -> dst). sumU: both directions.
__global__ void scatter0(const float* __restrict__ x, const int* __restrict__ src,
                         const int* __restrict__ dst,
                         float* __restrict__ sumO, float* __restrict__ sumU) {
  int t = blockIdx.x * blockDim.x + threadIdx.x;
  if (t >= NEDGES * 16) return;
  int e = t >> 4;
  int q = (t & 15) << 2;
  int s = src[e], d = dst[e];
  float4 xs = *(const float4*)(x + (size_t)s * 64 + q);
  float4 xd = *(const float4*)(x + (size_t)d * 64 + q);
  float* pO = sumO + (size_t)d * 64 + q;
  atomicAdd(pO + 0, xs.x); atomicAdd(pO + 1, xs.y); atomicAdd(pO + 2, xs.z); atomicAdd(pO + 3, xs.w);
  float* pUd = sumU + (size_t)d * 64 + q;
  atomicAdd(pUd + 0, xs.x); atomicAdd(pUd + 1, xs.y); atomicAdd(pUd + 2, xs.z); atomicAdd(pUd + 3, xs.w);
  float* pUs = sumU + (size_t)s * 64 + q;
  atomicAdd(pUs + 0, xd.x); atomicAdd(pUs + 1, xd.y); atomicAdd(pUs + 2, xd.z); atomicAdd(pUs + 3, xd.w);
}

// Layer-1 scatter: h1_* are [N,128] f16. sumI: 'I' (h1_0[dst] -> src). sumU: both dirs of h1_1.
__global__ void scatter1(const _Float16* __restrict__ h1_0, const _Float16* __restrict__ h1_1,
                         const int* __restrict__ src, const int* __restrict__ dst,
                         float* __restrict__ sumI, float* __restrict__ sumU) {
  int t = blockIdx.x * blockDim.x + threadIdx.x;
  if (t >= NEDGES * 32) return;
  int e = t >> 5;
  int q = (t & 31) << 2;
  int s = src[e], d = dst[e];
  const _Float16* p0 = h1_0 + (size_t)d * 128 + q;
  float* tI = sumI + (size_t)s * 128 + q;
  atomicAdd(tI + 0, (float)p0[0]); atomicAdd(tI + 1, (float)p0[1]);
  atomicAdd(tI + 2, (float)p0[2]); atomicAdd(tI + 3, (float)p0[3]);
  const _Float16* ps = h1_1 + (size_t)s * 128 + q;
  float* tUd = sumU + (size_t)d * 128 + q;
  atomicAdd(tUd + 0, (float)ps[0]); atomicAdd(tUd + 1, (float)ps[1]);
  atomicAdd(tUd + 2, (float)ps[2]); atomicAdd(tUd + 3, (float)ps[3]);
  const _Float16* pd = h1_1 + (size_t)d * 128 + q;
  float* tUs = sumU + (size_t)s * 128 + q;
  atomicAdd(tUs + 0, (float)pd[0]); atomicAdd(tUs + 1, (float)pd[1]);
  atomicAdd(tUs + 2, (float)pd[2]); atomicAdd(tUs + 3, (float)pd[3]);
}

// ---------------- fused SAGE layer GEMMs (WMMA) ----------------
// Layer 0: h1 = relu(x @ Ws0^T + mean @ Wn0^T + b0), K=64, out f16 [N,128]
__global__ __launch_bounds__(32) void sage_gemm0(
    const float* __restrict__ x,
    const float* __restrict__ sumO, const float* __restrict__ sumU,
    const float* __restrict__ degO, const float* __restrict__ degI,
    const _Float16* __restrict__ Ws0h, const _Float16* __restrict__ Wn0h,
    const float* __restrict__ b0,
    _Float16* __restrict__ h1_0, _Float16* __restrict__ h1_1) {
  const int head = blockIdx.y;
  const int row0 = blockIdx.x * 16;
  const int lane = threadIdx.x;
  const int m = lane & 15, hi = lane >> 4;
  const int node = row0 + m;

  const float* sums = head ? sumU : sumO;
  float deg = head ? (degO[node] + degI[node]) : degO[node];
  float scale = deg > 0.f ? 1.f / deg : 0.f;
  const _Float16* Ws = Ws0h + head * 128 * 64;
  const _Float16* Wn = Wn0h + head * 128 * 64;
  const float* bias = b0 + head * 128;
  _Float16* out = head ? h1_1 : h1_0;

  v16h ax[2], am[2];
#pragma unroll
  for (int kb = 0; kb < 2; ++kb) {
    ax[kb] = load_a_f32(x + (size_t)node * 64, kb, hi, 1.f);
    am[kb] = load_a_f32(sums + (size_t)node * 64, kb, hi, scale);
  }
#pragma unroll
  for (int ct = 0; ct < 8; ++ct) {
    int j = ct * 16 + m;
    float bv = bias[j];
    v8f c;
#pragma unroll
    for (int r = 0; r < 8; ++r) c[r] = bv;
#pragma unroll
    for (int kb = 0; kb < 2; ++kb) {
      v16h bS = load_b16(Ws, j, 64, kb, hi);
      c = __builtin_amdgcn_wmma_f32_16x16x32_f16(false, ax[kb], false, bS, (short)0, c, false, false);
      v16h bN = load_b16(Wn, j, 64, kb, hi);
      c = __builtin_amdgcn_wmma_f32_16x16x32_f16(false, am[kb], false, bN, (short)0, c, false, false);
    }
#pragma unroll
    for (int r = 0; r < 8; ++r) {
      float v = c[r];
      v = v > 0.f ? v : 0.f;                                   // relu
      out[(size_t)(row0 + r + hi * 8) * 128 + j] = (_Float16)v;
    }
  }
}

// Layer 1: out = h1 @ Ws1^T + mean1 @ Wn1^T + b1, K=128, out f32 [N,256] (head offset)
__global__ __launch_bounds__(32) void sage_gemm1(
    const _Float16* __restrict__ h1_0, const _Float16* __restrict__ h1_1,
    const float* __restrict__ sumI, const float* __restrict__ sumU,
    const float* __restrict__ degO, const float* __restrict__ degI,
    const _Float16* __restrict__ Ws1h, const _Float16* __restrict__ Wn1h,
    const float* __restrict__ b1,
    float* __restrict__ outbuf) {
  const int head = blockIdx.y;
  const int row0 = blockIdx.x * 16;
  const int lane = threadIdx.x;
  const int m = lane & 15, hi = lane >> 4;
  const int node = row0 + m;

  const _Float16* h1 = head ? h1_1 : h1_0;
  const float* sums = head ? sumU : sumI;
  float deg = head ? (degO[node] + degI[node]) : degI[node];
  float scale = deg > 0.f ? 1.f / deg : 0.f;
  const _Float16* Ws = Ws1h + head * 128 * 128;
  const _Float16* Wn = Wn1h + head * 128 * 128;
  const float* bias = b1 + head * 128;

  v16h ax[4], am[4];
#pragma unroll
  for (int kb = 0; kb < 4; ++kb) {
    ax[kb] = load_a_f16(h1 + (size_t)node * 128, kb, hi);
    am[kb] = load_a_f32(sums + (size_t)node * 128, kb, hi, scale);
  }
#pragma unroll
  for (int ct = 0; ct < 8; ++ct) {
    int j = ct * 16 + m;
    float bv = bias[j];
    v8f c;
#pragma unroll
    for (int r = 0; r < 8; ++r) c[r] = bv;
#pragma unroll
    for (int kb = 0; kb < 4; ++kb) {
      v16h bS = load_b16(Ws, j, 128, kb, hi);
      c = __builtin_amdgcn_wmma_f32_16x16x32_f16(false, ax[kb], false, bS, (short)0, c, false, false);
      v16h bN = load_b16(Wn, j, 128, kb, hi);
      c = __builtin_amdgcn_wmma_f32_16x16x32_f16(false, am[kb], false, bN, (short)0, c, false, false);
    }
#pragma unroll
    for (int r = 0; r < 8; ++r)
      outbuf[(size_t)(row0 + r + hi * 8) * 256 + head * 128 + j] = c[r];
  }
}

// ---------------- host launcher ----------------
extern "C" void kernel_launch(void* const* d_in, const int* in_sizes, int n_in,
                              void* d_out, int out_size, void* d_ws, size_t ws_size,
                              hipStream_t stream) {
  const float* x   = (const float*)d_in[0];
  const int*   src = (const int*)d_in[1];
  const int*   dst = (const int*)d_in[2];
  const float* Wn0 = (const float*)d_in[3];
  const float* Ws0 = (const float*)d_in[4];
  const float* b0  = (const float*)d_in[5];
  const float* Wn1 = (const float*)d_in[6];
  const float* Ws1 = (const float*)d_in[7];
  const float* b1  = (const float*)d_in[8];
  float* out = (float*)d_out;

  // workspace layout (256B-aligned regions)
  const size_t DEG_BYTES  = 400128;                 // N*4 padded
  const size_t SUM_BYTES  = (size_t)NNODES * 128 * 4;  // 51,200,000
  const size_t H1_BYTES   = (size_t)NNODES * 128 * 2;  // 25,600,000
  char* w = (char*)d_ws;
  float*    degO = (float*)(w);
  float*    degI = (float*)(w + DEG_BYTES);
  float*    sumA = (float*)(w + 2 * DEG_BYTES);
  float*    sumB = (float*)(w + 2 * DEG_BYTES + SUM_BYTES);
  _Float16* h1_0 = (_Float16*)(w + 2 * DEG_BYTES + 2 * SUM_BYTES);
  _Float16* h1_1 = (_Float16*)(w + 2 * DEG_BYTES + 2 * SUM_BYTES + H1_BYTES);
  _Float16* w16  = (_Float16*)(w + 2 * DEG_BYTES + 2 * SUM_BYTES + 2 * H1_BYTES);
  _Float16* Ws0h = w16;
  _Float16* Wn0h = w16 + 16384;
  _Float16* Ws1h = w16 + 32768;
  _Float16* Wn1h = w16 + 65536;

  // 1. zero deg + both sum buffers (contiguous)
  {
    int n4 = (int)((2 * DEG_BYTES + 2 * SUM_BYTES) / 16);
    zero_f4<<<(n4 + 255) / 256, 256, 0, stream>>>((float4*)w, n4);
  }
  // 2. weights -> f16
  convert_w<<<(98304 + 255) / 256, 256, 0, stream>>>(Wn0, Ws0, Wn1, Ws1, w16);
  // 3. degrees
  deg_kernel<<<(NEDGES + 255) / 256, 256, 0, stream>>>(src, dst, degO, degI);
  // 4. layer-0 scatter (ld 64)
  scatter0<<<(NEDGES * 16) / 256, 256, 0, stream>>>(x, src, dst, sumA, sumB);
  // 5. layer-0 fused GEMM (WMMA)
  sage_gemm0<<<dim3(NNODES / 16, 2), 32, 0, stream>>>(x, sumA, sumB, degO, degI,
                                                      Ws0h, Wn0h, b0, h1_0, h1_1);
  // 6. re-zero sums for layer 1
  {
    int n4 = (int)((2 * SUM_BYTES) / 16);
    zero_f4<<<(n4 + 255) / 256, 256, 0, stream>>>((float4*)sumA, n4);
  }
  // 7. layer-1 scatter (ld 128)
  scatter1<<<(NEDGES * 32) / 256, 256, 0, stream>>>(h1_0, h1_1, src, dst, sumA, sumB);
  // 8. layer-1 fused GEMM (WMMA) -> d_out [N,256]
  sage_gemm1<<<dim3(NNODES / 16, 2), 32, 0, stream>>>(h1_0, h1_1, sumA, sumB, degO, degI,
                                                      Ws1h, Wn1h, b1, out);
}
